// RandomGraphMixer3D_54073638256833
// MI455X (gfx1250) — matrix operand order repaired and checked by reading, not compiled, not run on hardware
//
#include <hip/hip_runtime.h>
#include <hip/hip_fp16.h>

typedef __attribute__((ext_vector_type(16))) _Float16 v16h;
typedef __attribute__((ext_vector_type(8)))  _Float16 v8h;
typedef __attribute__((ext_vector_type(8)))  float    v8f;

#define NNODES 32768   // X*Y*Z
#define BTDIM  16
#define CIN    32
#define COUT   32
#define RRAND  16
#define BCDIM  512     // BT*CIN

// ---------------------------------------------------------------------------
// Phase 1: transpose x viewed as (bc=512) x (n=32768) fp32, convert to f16,
// store xt[n][bc].  Classic LDS-tiled transpose; pad to 34 halves per row so
// the column-read stride is 68 B -> bank stride 17 (gcd(17,64)=1, conflict
// free on the 64-bank LDS).
// ---------------------------------------------------------------------------
__global__ __launch_bounds__(256)
void transpose_x_to_f16(const float* __restrict__ x, _Float16* __restrict__ xt) {
  __shared__ _Float16 tile[32][34];
  const int tx = threadIdx.x;          // 0..31 : n within tile (load), bc within tile (store)
  const int ty = threadIdx.y;          // 0..7
  const int n0  = blockIdx.x * 32;
  const int bc0 = blockIdx.y * 32;
#pragma unroll
  for (int k = 0; k < 4; ++k) {
    const int bcl = ty + 8 * k;
    tile[bcl][tx] = (_Float16)x[(size_t)(bc0 + bcl) * NNODES + (n0 + tx)];
  }
  __syncthreads();
#pragma unroll
  for (int k = 0; k < 4; ++k) {
    const int nl = ty + 8 * k;
    xt[(size_t)(n0 + nl) * BCDIM + (bc0 + tx)] = tile[tx][nl];
  }
}

// ---------------------------------------------------------------------------
// Phase 2: one wave32 computes out[b, 0..31, n0..n0+15] for one (b, node-tile)
// task via 16 K-chunks of v_wmma_f32_16x16x32_f16 per output o-tile.
//
//   D(o,n) = sum_r sum_c A_r(o,c) * B_r(c,n)
//   A_r(o,c) = weight[o][c][r]           (f16, from LDS, layout [r][o][c])
//   B_r(c,n) = xt[idx[n][r]][b*32 + c]   (f16 gather, contiguous per column)
//
// VGPR layouts assumed per ISA tables (wave32):
//   A 16x32 f16 : lane L holds row M=L%16; lanes<16 K=0-7 & 16-23,
//                 lanes>=16 K=8-15 & 24-31 (two 16B chunks at +0 and +16h).
//   B 32x16 f16 : lane L holds col N=L%16; lanes<16 K=0-7 & 16-23, lanes>=16 K=8-15 & 24-31
//                 (two 16B chunks at +0 and +16h from the K base).
//   D 16x16 f32 : VGPR g, lanes<16: (M=g, N=lane); lanes>=16: (M=g+8, N=lane-16).
// ---------------------------------------------------------------------------
__global__ __launch_bounds__(256)
void mixer_wmma(const _Float16* __restrict__ xt,
                const float*    __restrict__ weight,
                const float*    __restrict__ bias,
                const int*      __restrict__ idx,
                float*          __restrict__ out) {
  __shared__ _Float16 wlds[RRAND * COUT * CIN];   // [r][o][c], 32 KB

  // Cooperative convert weight[o][c][r] -> wlds[r*1024 + o*32 + c].
  for (int t = threadIdx.x; t < RRAND * COUT * CIN; t += 256) {
    const int r = t >> 10;
    const int o = (t >> 5) & 31;
    const int c = t & 31;
    wlds[t] = (_Float16)weight[(o * CIN + c) * RRAND + r];
  }
  __syncthreads();

  const int lane = threadIdx.x & 31;
  const int wid  = threadIdx.x >> 5;
  const int task = blockIdx.x * 8 + wid;         // 32768 tasks total
  const int b    = task >> 11;                   // / 2048
  const int n0   = (task & 2047) << 4;           // * 16
  const int nl   = lane & 15;                    // column within tile
  const int half = lane >> 4;                    // 0 or 1

  // Preload this lane's 16 gather indices (64B contiguous per node row).
  const int4* ip = (const int4*)(idx + (size_t)(n0 + nl) * RRAND);
  int j[16];
  { int4 q0 = ip[0], q1 = ip[1], q2 = ip[2], q3 = ip[3];
    j[0]=q0.x; j[1]=q0.y; j[2]=q0.z; j[3]=q0.w;
    j[4]=q1.x; j[5]=q1.y; j[6]=q1.z; j[7]=q1.w;
    j[8]=q2.x; j[9]=q2.y; j[10]=q2.z; j[11]=q2.w;
    j[12]=q3.x; j[13]=q3.y; j[14]=q3.z; j[15]=q3.w; }

  const _Float16* xb = xt + b * CIN + half * 8;    // K base within a node row
  v8f acc0 = {};   // o = 0..15 tile
  v8f acc1 = {};   // o = 16..31 tile

#pragma unroll
  for (int r = 0; r < RRAND; ++r) {
    // B gather: two 16B loads from this lane's node row.
    const _Float16* bp = xb + (size_t)j[r] * BCDIM;
    v8h blo = *(const v8h*)(bp);
    v8h bhi = *(const v8h*)(bp + 16);
    v16h B = __builtin_shufflevector(blo, bhi, 0,1,2,3,4,5,6,7,8,9,10,11,12,13,14,15);

    // A from LDS: lane row o = nl (+16 for second tile), K chunks at +0/+16h.
    const _Float16* ap = &wlds[r * (COUT * CIN) + nl * CIN + half * 8];
    v8h a0l = *(const v8h*)(ap);
    v8h a0h = *(const v8h*)(ap + 16);
    v16h A0 = __builtin_shufflevector(a0l, a0h, 0,1,2,3,4,5,6,7,8,9,10,11,12,13,14,15);
    v8h a1l = *(const v8h*)(ap + 16 * CIN);
    v8h a1h = *(const v8h*)(ap + 16 * CIN + 16);
    v16h A1 = __builtin_shufflevector(a1l, a1h, 0,1,2,3,4,5,6,7,8,9,10,11,12,13,14,15);

    acc0 = __builtin_amdgcn_wmma_f32_16x16x32_f16(false, A0, false, B,
                                                  (short)0, acc0, false, false);
    acc1 = __builtin_amdgcn_wmma_f32_16x16x32_f16(false, A1, false, B,
                                                  (short)0, acc1, false, false);
  }

  // Store D tiles with fused bias: per VGPR g, lanes 0-15 and 16-31 each
  // write one 64B-contiguous row segment of out[b][o][n0..n0+15].
  const int ncol = n0 + nl;
#pragma unroll
  for (int g = 0; g < 8; ++g) {
    const int o0 = g + half * 8;
    out[(size_t)(b * COUT + o0) * NNODES + ncol] = acc0[g] + bias[o0];
    const int o1 = o0 + 16;
    out[(size_t)(b * COUT + o1) * NNODES + ncol] = acc1[g] + bias[o1];
  }
}

// ---------------------------------------------------------------------------
// Fallback (workspace too small): one thread per (b, n), 32 fp32 accumulators.
// ---------------------------------------------------------------------------
__global__ __launch_bounds__(256)
void mixer_naive(const float* __restrict__ x, const float* __restrict__ w,
                 const float* __restrict__ bias, const int* __restrict__ idx,
                 float* __restrict__ out) {
  const size_t t = (size_t)blockIdx.x * blockDim.x + threadIdx.x;
  const int b = (int)(t >> 15);
  const int n = (int)(t & (NNODES - 1));
  float acc[COUT];
#pragma unroll
  for (int o = 0; o < COUT; ++o) acc[o] = bias[o];
  for (int r = 0; r < RRAND; ++r) {
    const int jn = idx[n * RRAND + r];
    for (int c = 0; c < CIN; ++c) {
      const float xv = x[((size_t)(b * CIN + c) << 15) + jn];
#pragma unroll
      for (int o = 0; o < COUT; ++o)
        acc[o] += w[(o * CIN + c) * RRAND + r] * xv;
    }
  }
#pragma unroll
  for (int o = 0; o < COUT; ++o)
    out[(size_t)(b * COUT + o) * NNODES + n] = acc[o];
}

// ---------------------------------------------------------------------------
extern "C" void kernel_launch(void* const* d_in, const int* in_sizes, int n_in,
                              void* d_out, int out_size, void* d_ws, size_t ws_size,
                              hipStream_t stream) {
  const float* x      = (const float*)d_in[0];   // (16, 32, 32768) fp32
  const float* weight = (const float*)d_in[1];   // (32, 32, 16)    fp32
  const float* bias   = (const float*)d_in[2];   // (32,)           fp32
  const int*   ridx   = (const int*)d_in[3];     // (32768, 16)     int32
  float* out = (float*)d_out;                    // (16, 32, 32768) fp32

  const size_t need = (size_t)NNODES * BCDIM * sizeof(_Float16);  // 32 MB
  if (d_ws && ws_size >= need) {
    _Float16* xt = (_Float16*)d_ws;
    // Phase 1: transpose + fp32->fp16 convert. Grid covers 512 x 32768.
    transpose_x_to_f16<<<dim3(NNODES / 32, BCDIM / 32), dim3(32, 8), 0, stream>>>(x, xt);
    // Phase 2: 32768 wave-tasks, 8 waves (256 threads) per block.
    mixer_wmma<<<dim3((BTDIM * (NNODES / 16)) / 8), dim3(256), 0, stream>>>(
        xt, weight, bias, ridx, out);
  } else {
    mixer_naive<<<dim3((BTDIM * NNODES) / 256), dim3(256), 0, stream>>>(
        x, weight, bias, ridx, out);
  }
}
